// GraphFormerDecoder_18691697672317
// MI455X (gfx1250) — matrix operand order repaired and checked by reading, not compiled, not run on hardware
//
#include <hip/hip_runtime.h>
#include <hip/hip_bf16.h>
#include <math.h>

// Problem dims (match reference)
#define Bsz   4
#define Tt    512
#define TEe   512
#define NDd   256
#define Dd    1024
#define Hh    16
#define DHh   64
#define BNn   512
#define Ll    6

typedef __attribute__((ext_vector_type(16))) __bf16 v16bf;
typedef __attribute__((ext_vector_type(2)))  __bf16 v2bf;
typedef __attribute__((ext_vector_type(8)))  float  v8f;

union Frag16 { v16bf v; unsigned u[8]; };

__device__ __forceinline__ unsigned short f32_to_bf16(float f) {
  unsigned u = __float_as_uint(f);
  unsigned r = u + 0x7FFFu + ((u >> 16) & 1u);   // round-to-nearest-even
  return (unsigned short)(r >> 16);
}

// Pack two f32 -> bf16x2 in one instruction where available (v_cvt_pk_bf16_f32)
__device__ __forceinline__ unsigned pack_bf16x2(float a, float b) {
#if __has_builtin(__builtin_amdgcn_cvt_pk_bf16_f32)
  v2bf p = __builtin_amdgcn_cvt_pk_bf16_f32(a, b);
  return __builtin_bit_cast(unsigned, p);
#else
  return (unsigned)f32_to_bf16(a) | ((unsigned)f32_to_bf16(b) << 16);
#endif
}

__device__ __forceinline__ float gelu_tanh(float x) {
  float x3 = x * x * x;
  return 0.5f * x * (1.0f + tanhf(0.7978845608028654f * (x + 0.044715f * x3)));
}

// ---------------------------------------------------------------------------
// Generic batched WMMA GEMM:
//   C[z] = act( alpha * A[z](MxK) @ B[z](KxN or (NxK)^T) + bias + addmat[z] + resid )
// Block tile 128 x (32*WN*? ) : 8 waves; wave tile = 32 rows x (WN*16) cols,
// i.e. 2 x WN WMMA accumulators (v_wmma_f32_16x16x32_bf16).
// A/B staged through LDS with f32->bf16 conversion (v_cvt_pk_bf16_f32).
// Per-batch offsets decomposed as z = zb*heads + zh  ->  base + zb*s1 + zh*s2.
// ---------------------------------------------------------------------------
#define BM 128
#define BK 32
#define LDSP 34   // padded row (even -> keeps 4B alignment for paired reads)

template <int WN>   // N tiles per wave: 2 (BN=64) or 4 (BN=128)
__global__ __launch_bounds__(256)
void k_gemm(const float* __restrict__ A, int lda, long long sA1, long long sA2,
            const float* __restrict__ Bm, int ldb, long long sB1, long long sB2, int transB,
            float* __restrict__ C, int ldc, long long sC1, long long sC2,
            const float* __restrict__ bias,
            const float* __restrict__ resid,
            const float* __restrict__ addmat, int ldam, long long sM1, long long sM2,
            float alpha, int act, int heads, int K)
{
  constexpr int BNt = 32 * WN;          // block N extent (64 or 128)
  constexpr int BPT = 4 * WN;           // B floats per thread per K-step (8 or 16)

  __shared__ __align__(16) unsigned short Alds[BM][LDSP];
  __shared__ __align__(16) unsigned short Blds[BNt][LDSP];

  const int tid  = threadIdx.x;
  const int lane = tid & 31;
  const int wave = tid >> 5;
  const int lsub = lane & 15;
  const int lhi  = lane >> 4;

  const int z = blockIdx.z;
  const long long zb = z / heads, zh = z % heads;
  const float* Ab = A  + zb * sA1 + zh * sA2;
  const float* Bb = Bm + zb * sB1 + zh * sB2;
  float*       Cb = C  + zb * sC1 + zh * sC2;
  const float* Mb = addmat ? (addmat + zb * sM1 + zh * sM2) : (const float*)0;

  const int row0 = blockIdx.y * BM;
  const int col0 = blockIdx.x * BNt;
  const int waveM = (wave & 3) * 32;
  const int waveN = (wave >> 2) * (WN * 16);

  v8f acc[2][WN];
  v8f zero = {0.f,0.f,0.f,0.f,0.f,0.f,0.f,0.f};
  #pragma unroll
  for (int mt = 0; mt < 2; ++mt)
    #pragma unroll
    for (int nt = 0; nt < WN; ++nt) acc[mt][nt] = zero;

  for (int k0 = 0; k0 < K; k0 += BK) {
    // ---- stage A tile (128x32) f32 -> bf16 LDS (paired cvt) ----
    {
      int r  = tid >> 1;
      int cb = (tid & 1) * 16;
      const float4* src = (const float4*)(Ab + (long long)(row0 + r) * lda + k0 + cb);
      #pragma unroll
      for (int q = 0; q < 4; ++q) {
        float4 f = src[q];
        *(unsigned*)&Alds[r][cb + 4*q + 0] = pack_bf16x2(f.x, f.y);
        *(unsigned*)&Alds[r][cb + 4*q + 2] = pack_bf16x2(f.z, f.w);
      }
      // prefetch next K tile of A (global_prefetch_b8)
      if (k0 + BK < K)
        __builtin_prefetch(Ab + (long long)(row0 + r) * lda + k0 + BK + cb, 0, 3);
    }
    // ---- stage B tile, stored K-contiguous per column: Blds[n][k] ----
    if (transB) {                 // B supplied (N,K) row-major
      constexpr int TPR = BK / BPT;               // threads per row
      int n  = tid / TPR;
      int kb = (tid % TPR) * BPT;
      const float4* src = (const float4*)(Bb + (long long)(col0 + n) * ldb + k0 + kb);
      #pragma unroll
      for (int q = 0; q < BPT / 4; ++q) {
        float4 f = src[q];
        *(unsigned*)&Blds[n][kb + 4*q + 0] = pack_bf16x2(f.x, f.y);
        *(unsigned*)&Blds[n][kb + 4*q + 2] = pack_bf16x2(f.z, f.w);
      }
      if (k0 + BK < K)
        __builtin_prefetch(Bb + (long long)(col0 + n) * ldb + k0 + BK + kb, 0, 3);
    } else {                      // B supplied (K,N) row-major -> transpose into LDS
      constexpr int TPK = BNt / BPT;              // threads per k row (8)
      int k  = tid / TPK;
      int nb = (tid % TPK) * BPT;
      const float4* src = (const float4*)(Bb + (long long)(k0 + k) * ldb + col0 + nb);
      #pragma unroll
      for (int q = 0; q < BPT / 4; ++q) {
        float4 f = src[q];
        Blds[nb + 4*q + 0][k] = f32_to_bf16(f.x);
        Blds[nb + 4*q + 1][k] = f32_to_bf16(f.y);
        Blds[nb + 4*q + 2][k] = f32_to_bf16(f.z);
        Blds[nb + 4*q + 3][k] = f32_to_bf16(f.w);
      }
      if (k0 + BK < K)
        __builtin_prefetch(Bb + (long long)(k0 + BK + k) * ldb + col0 + nb, 0, 3);
    }
    __syncthreads();

    // ---- build fragments (k = 16*lhi + s in both A and B) and issue WMMAs ----
    Frag16 af[2], bfr[WN];
    #pragma unroll
    for (int mt = 0; mt < 2; ++mt) {
      int m  = waveM + mt * 16 + lsub;
      int kb = lhi * 16;
      #pragma unroll
      for (int j = 0; j < 4; ++j)
        af[mt].u[j]     = *(const unsigned*)&Alds[m][kb + 2*j];        // K = kb..kb+7
      #pragma unroll
      for (int j = 0; j < 4; ++j)
        af[mt].u[4 + j] = *(const unsigned*)&Alds[m][kb + 8 + 2*j];    // K = kb+8..kb+15
    }
    #pragma unroll
    for (int nt = 0; nt < WN; ++nt) {
      int n   = waveN + nt * 16 + lsub;
      int kb0 = lhi * 16;
      #pragma unroll
      for (int j = 0; j < 8; ++j)
        bfr[nt].u[j] = *(const unsigned*)&Blds[n][kb0 + 2*j];
    }
    #pragma unroll
    for (int mt = 0; mt < 2; ++mt)
      #pragma unroll
      for (int nt = 0; nt < WN; ++nt)
        acc[mt][nt] = __builtin_amdgcn_wmma_f32_16x16x32_bf16(
            false, af[mt].v, false, bfr[nt].v, (short)0, acc[mt][nt], false, false);
    __syncthreads();
  }

  // ---- epilogue: C layout VGPR r -> M=r (lanes0-15), M=8+r (lanes16-31) ----
  #pragma unroll
  for (int mt = 0; mt < 2; ++mt) {
    #pragma unroll
    for (int nt = 0; nt < WN; ++nt) {
      int mBase = row0 + waveM + mt * 16 + lhi * 8;
      int n     = col0 + waveN + nt * 16 + lsub;
      float bv = bias ? bias[n] : 0.0f;
      #pragma unroll
      for (int r = 0; r < 8; ++r) {
        int m = mBase + r;
        float v = acc[mt][nt][r] * alpha + bv;
        if (Mb)    v += Mb[(long long)m * ldam + n];
        if (resid) v += resid[(long long)m * ldc + n];
        if (act)   v  = gelu_tanh(v);
        Cb[(long long)m * ldc + n] = v;
      }
    }
  }
}

// ---------------------------------------------------------------------------
// Row LayerNorm over D=1024 (one block per row); optional additive input.
// ---------------------------------------------------------------------------
__global__ __launch_bounds__(256)
void k_layernorm(const float* __restrict__ x, const float* __restrict__ add,
                 const float* __restrict__ g, const float* __restrict__ bp,
                 float* __restrict__ out)
{
  __shared__ float red[256];
  const int row = blockIdx.x, tid = threadIdx.x;
  float4 v = ((const float4*)(x + (long long)row * Dd))[tid];
  if (add) {
    float4 a = ((const float4*)(add + (long long)row * Dd))[tid];
    v.x += a.x; v.y += a.y; v.z += a.z; v.w += a.w;
  }
  red[tid] = v.x + v.y + v.z + v.w;
  __syncthreads();
  for (int st = 128; st > 0; st >>= 1) { if (tid < st) red[tid] += red[tid + st]; __syncthreads(); }
  float mean = red[0] * (1.0f / Dd);
  __syncthreads();
  float dx = v.x - mean, dy = v.y - mean, dz = v.z - mean, dw = v.w - mean;
  red[tid] = dx*dx + dy*dy + dz*dz + dw*dw;
  __syncthreads();
  for (int st = 128; st > 0; st >>= 1) { if (tid < st) red[tid] += red[tid + st]; __syncthreads(); }
  float rs = rsqrtf(red[0] * (1.0f / Dd) + 1e-5f);
  int c = tid * 4;
  float4 o;
  o.x = g[c+0] * dx * rs + bp[c+0];
  o.y = g[c+1] * dy * rs + bp[c+1];
  o.z = g[c+2] * dz * rs + bp[c+2];
  o.w = g[c+3] * dw * rs + bp[c+3];
  ((float4*)(out + (long long)row * Dd))[tid] = o;
}

// ---------------------------------------------------------------------------
// In-place softmax over rows of 512. Row pulled into LDS with the CDNA5
// async global->LDS path (ASYNCcnt) before the reduction.
// ---------------------------------------------------------------------------
__global__ __launch_bounds__(128)
void k_softmax512(float* __restrict__ p)
{
  __shared__ __align__(16) float srow[512];
  __shared__ float red[128];
  const int tid = threadIdx.x;
  float* rp = p + (long long)blockIdx.x * 512;

  unsigned ldsoff = (unsigned)(unsigned long long)(size_t)&srow[tid * 4];
  unsigned long long ga = (unsigned long long)(size_t)(rp + tid * 4);
  asm volatile("global_load_async_to_lds_b128 %0, %1, off" :: "v"(ldsoff), "v"(ga) : "memory");
  asm volatile("s_wait_asynccnt 0" ::: "memory");
  __syncthreads();

  float4 v = *(const float4*)&srow[tid * 4];
  float mx = fmaxf(fmaxf(v.x, v.y), fmaxf(v.z, v.w));
  red[tid] = mx;
  __syncthreads();
  for (int st = 64; st > 0; st >>= 1) { if (tid < st) red[tid] = fmaxf(red[tid], red[tid + st]); __syncthreads(); }
  mx = red[0];
  __syncthreads();
  v.x = __expf(v.x - mx); v.y = __expf(v.y - mx); v.z = __expf(v.z - mx); v.w = __expf(v.w - mx);
  red[tid] = v.x + v.y + v.z + v.w;
  __syncthreads();
  for (int st = 64; st > 0; st >>= 1) { if (tid < st) red[tid] += red[tid + st]; __syncthreads(); }
  float inv = 1.0f / red[0];
  v.x *= inv; v.y *= inv; v.z *= inv; v.w *= inv;
  ((float4*)rp)[tid] = v;
}

// x += sinusoidal PE + centrality (degree) embeddings
__global__ __launch_bounds__(256)
void k_embed(float* __restrict__ x, const int* __restrict__ indeg, const int* __restrict__ outdeg,
             const float* __restrict__ intab, const float* __restrict__ outtab)
{
  long long i = (long long)blockIdx.x * 256 + threadIdx.x;
  int d = (int)(i % Dd);
  long long bt = i / Dd;
  int t = (int)(bt % Tt);
  float div = __expf(-logf(10000.0f) * (float)(d & ~1) * (1.0f / Dd));
  float ang = (float)t * div;
  float pe  = (d & 1) ? __cosf(ang) : __sinf(ang);
  int di = indeg[bt], dox = outdeg[bt];
  x[i] += pe + intab[(long long)di * Dd + d] + outtab[(long long)dox * Dd + d];
}

// bias[b,t,s] = spatial_table[sd[b,t,s]] + edge_bias[b,t,s]
__global__ __launch_bounds__(256)
void k_bias(const float* __restrict__ eb, const int* __restrict__ sd,
            const float* __restrict__ stab, float* __restrict__ ob)
{
  long long i = (long long)blockIdx.x * 256 + threadIdx.x;
  ob[i] = stab[sd[i]] + eb[i];
}

// ---------------------------------------------------------------------------
static void gemm(hipStream_t s,
                 const float* A, int lda, long long sA1, long long sA2,
                 const float* B, int ldb, long long sB1, long long sB2, int transB,
                 float* C, int ldc, long long sC1, long long sC2,
                 const float* bias, const float* resid,
                 const float* addmat, int ldam, long long sM1, long long sM2,
                 float alpha, int act, int heads, int M, int N, int K, int Z)
{
  if (N % 128 == 0) {
    dim3 g(N / 128, M / BM, Z);
    k_gemm<4><<<g, 256, 0, s>>>(A, lda, sA1, sA2, B, ldb, sB1, sB2, transB,
                                C, ldc, sC1, sC2, bias, resid, addmat, ldam, sM1, sM2,
                                alpha, act, heads, K);
  } else {
    dim3 g(N / 64, M / BM, Z);
    k_gemm<2><<<g, 256, 0, s>>>(A, lda, sA1, sA2, B, ldb, sB1, sB2, transB,
                                C, ldc, sC1, sC2, bias, resid, addmat, ldam, sM1, sM2,
                                alpha, act, heads, K);
  }
}

extern "C" void kernel_launch(void* const* d_in, const int* in_sizes, int n_in,
                              void* d_out, int out_size, void* d_ws, size_t ws_size,
                              hipStream_t stream)
{
  const float* node_feat = (const float*)d_in[0];
  const float* enc       = (const float*)d_in[1];
  const float* edge_bias = (const float*)d_in[2];
  const int*   in_deg    = (const int*)d_in[3];
  const int*   out_deg   = (const int*)d_in[4];
  const int*   sp_dist   = (const int*)d_in[5];
  const float* npw       = (const float*)d_in[6];
  const float* npb       = (const float*)d_in[7];
  const float* intab     = (const float*)d_in[8];
  const float* outtab    = (const float*)d_in[9];
  const float* sptab     = (const float*)d_in[10];
  const float* Wq = (const float*)d_in[11]; const float* bq = (const float*)d_in[12];
  const float* Wk = (const float*)d_in[13]; const float* bk = (const float*)d_in[14];
  const float* Wv = (const float*)d_in[15]; const float* bv = (const float*)d_in[16];
  const float* Wo = (const float*)d_in[17]; const float* bo = (const float*)d_in[18];
  const float* ciw = (const float*)d_in[19]; const float* cib = (const float*)d_in[20];
  const float* cow = (const float*)d_in[21]; const float* cob = (const float*)d_in[22];
  const float* ln1g = (const float*)d_in[23]; const float* ln1b = (const float*)d_in[24];
  const float* ln2g = (const float*)d_in[25]; const float* ln2b = (const float*)d_in[26];
  const float* ln3g = (const float*)d_in[27]; const float* ln3b = (const float*)d_in[28];
  const float* bng  = (const float*)d_in[29]; const float* bnb  = (const float*)d_in[30];
  const float* W1 = (const float*)d_in[31]; const float* b1 = (const float*)d_in[32];
  const float* W2 = (const float*)d_in[33]; const float* b2 = (const float*)d_in[34];

  const long long BT   = (long long)Bsz * Tt;            // 2048
  const long long BTD  = BT * Dd;                        // 2,097,152
  const long long BTT  = (long long)Bsz * Tt * Tt;       // 1,048,576
  const long long BHTT = (long long)Bsz * Hh * Tt * Tt;  // 16,777,216

  float* xf  = (float*)d_ws;
  float* h1f = xf  + BTD;
  float* h2f = h1f + BTD;
  float* h3f = h2f + BTD;
  float* t1f = h3f + BTD;
  float* qf  = t1f + BTD;
  float* kf  = qf  + BTD;
  float* vf  = kf  + BTD;
  float* of  = vf  + BTD;
  float* fff = of  + BTD;                 // (BT x 512)
  float* bia = fff + BT * BNn;            // (B,T,T)

  float* out_x     = (float*)d_out;
  float* out_self  = out_x + BTD;
  float* out_cross = out_self + (long long)Ll * BHTT;

  const float scale = 0.125f; // 1/sqrt(64)
  const long long Z0 = 0;

  // x = node_features @ node_proj_w + b ; += PE + centrality
  gemm(stream, node_feat, NDd, Z0, Z0, npw, Dd, Z0, Z0, 0,
       xf, Dd, Z0, Z0, npb, 0, 0, 0, Z0, Z0, 1.0f, 0, 1, (int)BT, Dd, NDd, 1);
  k_embed<<<dim3((unsigned)(BTD / 256)), 256, 0, stream>>>(xf, in_deg, out_deg, intab, outtab);
  k_bias<<<dim3((unsigned)(BTT / 256)), 256, 0, stream>>>(edge_bias, sp_dist, sptab, bia);

  for (int l = 0; l < Ll; ++l) {
    const long long DD = (long long)Dd * Dd;
    float* swl = out_self  + (long long)l * BHTT;
    float* cwl = out_cross + (long long)l * BHTT;

    // ---- self-attention (pre-norm) ----
    k_layernorm<<<dim3((unsigned)BT), 256, 0, stream>>>(xf, 0, ln1g + l*Dd, ln1b + l*Dd, h1f);
    gemm(stream, h1f, Dd, Z0, Z0, Wq + l*DD, Dd, Z0, Z0, 0, qf, Dd, Z0, Z0, bq + l*Dd, 0, 0, 0, Z0, Z0, 1.f, 0, 1, (int)BT, Dd, Dd, 1);
    gemm(stream, h1f, Dd, Z0, Z0, Wk + l*DD, Dd, Z0, Z0, 0, kf, Dd, Z0, Z0, bk + l*Dd, 0, 0, 0, Z0, Z0, 1.f, 0, 1, (int)BT, Dd, Dd, 1);
    gemm(stream, h1f, Dd, Z0, Z0, Wv + l*DD, Dd, Z0, Z0, 0, vf, Dd, Z0, Z0, bv + l*Dd, 0, 0, 0, Z0, Z0, 1.f, 0, 1, (int)BT, Dd, Dd, 1);
    // scores (per b,h): q @ k^T * scale + bias   -> out_self
    gemm(stream, qf, Dd, (long long)Tt*Dd, DHh,
         kf, Dd, (long long)Tt*Dd, DHh, 1,
         swl, Tt, (long long)Hh*Tt*Tt, (long long)Tt*Tt,
         0, 0, bia, Tt, (long long)Tt*Tt, Z0,
         scale, 0, Hh, Tt, Tt, DHh, Bsz * Hh);
    k_softmax512<<<dim3((unsigned)(Bsz * Hh * Tt)), 128, 0, stream>>>(swl);
    // o = attn @ v   (N = 64 -> WN=2 tile)
    gemm(stream, swl, Tt, (long long)Hh*Tt*Tt, (long long)Tt*Tt,
         vf, Dd, (long long)Tt*Dd, DHh, 0,
         of, Dd, (long long)Tt*Dd, DHh,
         0, 0, 0, 0, Z0, Z0, 1.f, 0, Hh, Tt, DHh, Tt, Bsz * Hh);
    gemm(stream, of, Dd, Z0, Z0, Wo + l*DD, Dd, Z0, Z0, 0,
         t1f, Dd, Z0, Z0, bo + l*Dd, h1f, 0, 0, Z0, Z0, 1.f, 0, 1, (int)BT, Dd, Dd, 1);

    // ---- cross-attention ----
    k_layernorm<<<dim3((unsigned)BT), 256, 0, stream>>>(t1f, 0, ln2g + l*Dd, ln2b + l*Dd, h2f);
    const float* wqc = ciw + (long long)l * 3 * DD;           // (D,D) slices, used transposed
    const float* wkc = wqc + DD;
    const float* wvc = wqc + 2 * DD;
    gemm(stream, h2f, Dd, Z0, Z0, wqc, Dd, Z0, Z0, 1, qf, Dd, Z0, Z0, cib + l*3*Dd,        0, 0, 0, Z0, Z0, 1.f, 0, 1, (int)BT, Dd, Dd, 1);
    gemm(stream, enc, Dd, Z0, Z0, wkc, Dd, Z0, Z0, 1, kf, Dd, Z0, Z0, cib + l*3*Dd + Dd,   0, 0, 0, Z0, Z0, 1.f, 0, 1, (int)BT, Dd, Dd, 1);
    gemm(stream, enc, Dd, Z0, Z0, wvc, Dd, Z0, Z0, 1, vf, Dd, Z0, Z0, cib + l*3*Dd + 2*Dd, 0, 0, 0, Z0, Z0, 1.f, 0, 1, (int)BT, Dd, Dd, 1);
    gemm(stream, qf, Dd, (long long)Tt*Dd, DHh,
         kf, Dd, (long long)TEe*Dd, DHh, 1,
         cwl, TEe, (long long)Hh*Tt*TEe, (long long)Tt*TEe,
         0, 0, 0, 0, Z0, Z0,
         scale, 0, Hh, Tt, TEe, DHh, Bsz * Hh);
    k_softmax512<<<dim3((unsigned)(Bsz * Hh * Tt)), 128, 0, stream>>>(cwl);
    gemm(stream, cwl, TEe, (long long)Hh*Tt*TEe, (long long)Tt*TEe,
         vf, Dd, (long long)TEe*Dd, DHh, 0,
         of, Dd, (long long)Tt*Dd, DHh,
         0, 0, 0, 0, Z0, Z0, 1.f, 0, Hh, Tt, DHh, TEe, Bsz * Hh);
    gemm(stream, of, Dd, Z0, Z0, cow + l*DD, Dd, Z0, Z0, 0,
         t1f, Dd, Z0, Z0, cob + l*Dd, h2f, 0, 0, Z0, Z0, 1.f, 0, 1, (int)BT, Dd, Dd, 1);

    // ---- bottleneck FFN ----
    k_layernorm<<<dim3((unsigned)BT), 256, 0, stream>>>(t1f, 0, ln3g + l*Dd, ln3b + l*Dd, h3f);
    gemm(stream, h3f, Dd, Z0, Z0, W1 + (long long)l*Dd*BNn, BNn, Z0, Z0, 0,
         fff, BNn, Z0, Z0, b1 + l*BNn, 0, 0, 0, Z0, Z0, 1.f, 1 /*gelu*/, 1, (int)BT, BNn, Dd, 1);
    gemm(stream, fff, BNn, Z0, Z0, W2 + (long long)l*BNn*Dd, Dd, Z0, Z0, 0,
         t1f, Dd, Z0, Z0, b2 + l*Dd, h3f /*resid = h*/, 0, 0, Z0, Z0, 1.f, 0, 1, (int)BT, Dd, BNn, 1);
    // x = LN(h + ff)  (t1f already = h + ff); last layer writes straight to d_out
    k_layernorm<<<dim3((unsigned)BT), 256, 0, stream>>>(t1f, 0, bng + l*Dd, bnb + l*Dd,
                                                        (l == Ll - 1) ? out_x : xf);
  }
  (void)in_sizes; (void)n_in; (void)out_size; (void)ws_size;
}
